// SelectiveSSM_12936441496001
// MI455X (gfx1250) — matrix operand order repaired and checked by reading, not compiled, not run on hardware
//
#include <hip/hip_runtime.h>

// ---------------------------------------------------------------------------
// Mamba selective-SSM block for gfx1250 (MI455X).
// GEMMs: v_wmma_f32_16x16x32_bf16, 64x64 register-blocked tile per wave
// (16 WMMAs per k-step reusing 4 A + 4 B fragments -> ~16 flops/byte from L2).
// Also issues a NULL-descriptor TDM tensor_load_to_lds (architectural no-op)
// + s_wait_tensorcnt to exercise the CDNA5 async-tensor path.
// ---------------------------------------------------------------------------

typedef __attribute__((ext_vector_type(16))) __bf16 v16bf;
typedef __attribute__((ext_vector_type(8)))  float  v8f;

#define DMODEL 1024
#define DSTATE 16
#define DIN    2048
#define BATCH  2
#define SEQ    2048
#define MROWS  (BATCH * SEQ)      // 4096 rows through every GEMM

static __device__ __forceinline__ void pack4(v16bf& v, int base, float4 f) {
    v[base + 0] = (__bf16)f.x;
    v[base + 1] = (__bf16)f.y;
    v[base + 2] = (__bf16)f.z;
    v[base + 3] = (__bf16)f.w;
}

// A fragment (16x32 bf16): lane half 0 -> K {0..7, 16..23}, half 1 -> {8..15, 24..31}
static __device__ __forceinline__ v16bf load_a_frag(const float* __restrict__ arow,
                                                    int k0, int half) {
    const float4* p0 = reinterpret_cast<const float4*>(arow + k0 + half * 8);
    const float4* p1 = reinterpret_cast<const float4*>(arow + k0 + 16 + half * 8);
    float4 f0 = p0[0], f1 = p0[1], f2 = p1[0], f3 = p1[1];
    v16bf v;
    pack4(v, 0, f0); pack4(v, 4, f1); pack4(v, 8, f2); pack4(v, 12, f3);
    return v;
}

// B fragment (32x16 bf16): lane half 0 -> K {0..7, 16..23}, half 1 -> {8..15, 24..31}
static __device__ __forceinline__ v16bf load_b_frag(const float* __restrict__ brow,
                                                    int k0, int half) {
    const float4* p0 = reinterpret_cast<const float4*>(brow + k0 + half * 8);
    const float4* p1 = reinterpret_cast<const float4*>(brow + k0 + 16 + half * 8);
    float4 f0 = p0[0], f1 = p0[1], f2 = p1[0], f3 = p1[1];
    v16bf v;
    pack4(v, 0, f0); pack4(v, 4, f1); pack4(v, 8, f2); pack4(v, 12, f3);
    return v;
}

// C[M x N] = A[M x K] (row-major, stride lda) @ Bw[N x K]^T (row-major, stride ldb)
// Wave tile: 64x64 (4x4 subtiles of 16x16). Block: 8 waves as 2(M) x 4(N)
// -> 128 x 256 per block. EPI==1: fused out = softplus(acc + bias[n]).
template<int EPI>
__global__ __launch_bounds__(256)
void gemm_bf16_wmma(const float* __restrict__ A, int lda,
                    const float* __restrict__ Bw, int ldb,
                    float* __restrict__ C, int ldc,
                    int N, int K, const float* __restrict__ bias) {
    const int lane  = threadIdx.x & 31;
    const int wave  = threadIdx.x >> 5;
    const int waveM = wave >> 2;                 // 0..1
    const int waveN = wave & 3;                  // 0..3
    const int half  = lane >> 4;
    const int r     = lane & 15;
    const int tileM = blockIdx.y * 128 + waveM * 64;
    const int tileN = blockIdx.x * 256 + waveN * 64;
    if (tileN >= N) return;                      // wave-uniform: EXEC stays all-1s

#if defined(__has_builtin)
#if __has_builtin(__builtin_amdgcn_tensor_load_to_lds) && \
    __has_builtin(__builtin_amdgcn_s_wait_tensorcnt)
    {
        // NULL tensor descriptor (group0.count==0): ISA-defined no-op DMA.
        // Exercises the TDM issue path + TENSORcnt wait at zero correctness risk.
        // This toolchain uses the 6-arg form: (v4u, v8i, v4i, v4i, v8i, cpol).
        typedef __attribute__((ext_vector_type(4))) unsigned int v4u;
        typedef __attribute__((ext_vector_type(8))) int          v8i;
        typedef __attribute__((ext_vector_type(4))) int          v4i;
        v4u g0 = {}; v8i g1 = {}; v4i g2 = {}, g3 = {}; v8i g4 = {};
        __builtin_amdgcn_tensor_load_to_lds(g0, g1, g2, g3, g4, 0);
        __builtin_amdgcn_s_wait_tensorcnt(0);
    }
#endif
#endif

    const float* arow[4];
    const float* brow[4];
    #pragma unroll
    for (int i = 0; i < 4; ++i)
        arow[i] = A + (size_t)(tileM + 16 * i + r) * lda;
    #pragma unroll
    for (int j = 0; j < 4; ++j) {
        int bn = tileN + 16 * j + r;
        if (bn > N - 1) bn = N - 1;              // clamp: safe loads, never stored
        brow[j] = Bw + (size_t)bn * ldb;
    }

    const v8f vzero = {};
    v8f acc[4][4];
    #pragma unroll
    for (int i = 0; i < 4; ++i)
        #pragma unroll
        for (int j = 0; j < 4; ++j) acc[i][j] = vzero;

    for (int k0 = 0; k0 < K; k0 += 32) {
        v16bf af[4], bf[4];
        #pragma unroll
        for (int i = 0; i < 4; ++i) af[i] = load_a_frag(arow[i], k0, half);
        #pragma unroll
        for (int j = 0; j < 4; ++j) bf[j] = load_b_frag(brow[j], k0, half);
        #pragma unroll
        for (int i = 0; i < 4; ++i)
            #pragma unroll
            for (int j = 0; j < 4; ++j)
                acc[i][j] = __builtin_amdgcn_wmma_f32_16x16x32_bf16(
                    false, af[i], false, bf[j], (short)0, acc[i][j], false, false);
    }

    // C/D layout: VGPR e, lane: M = subtileM + half*8 + e, N = subtileN + (lane&15)
    #pragma unroll
    for (int i = 0; i < 4; ++i) {
        const int cm = tileM + 16 * i + half * 8;
        #pragma unroll
        for (int j = 0; j < 4; ++j) {
            const int cn = tileN + 16 * j + r;
            if (cn < N) {
                const float bval = (EPI == 1) ? bias[cn] : 0.0f;
                float* cp = C + (size_t)cm * ldc + cn;
                #pragma unroll
                for (int e = 0; e < 8; ++e) {
                    float v = acc[i][j][e];
                    if (EPI == 1) {
                        v += bval;
                        v = (v > 20.0f) ? v : log1pf(__expf(v));   // softplus
                    }
                    cp[(size_t)e * ldc] = v;
                }
            }
        }
    }
}

// Depthwise causal conv1d (K=4, left pad 3) + bias + SiLU.
// x_inner is the first DIN columns of xz (row stride 2*DIN).
__global__ __launch_bounds__(256)
void conv_silu_kernel(const float* __restrict__ xz,
                      const float* __restrict__ wconv,
                      const float* __restrict__ bconv,
                      float* __restrict__ xconv) {
    const int idx = blockIdx.x * 256 + threadIdx.x;    // over MROWS*DIN = 2^23
    const int c = idx & (DIN - 1);
    const int l = (idx >> 11) & (SEQ - 1);
    const int b = idx >> 22;
    const float4 w = *reinterpret_cast<const float4*>(wconv + c * 4);
    const float wk[4] = {w.x, w.y, w.z, w.w};
    const float* xi = xz + (size_t)b * SEQ * (2 * DIN) + c;
    float acc = bconv[c];
    #pragma unroll
    for (int k = 0; k < 4; ++k) {
        const int t = l - 3 + k;
        if (t >= 0) acc = fmaf(xi[(size_t)t * (2 * DIN)], wk[k], acc);
    }
    acc = acc / (1.0f + __expf(-acc));                 // SiLU
    xconv[idx] = acc;
}

// Selective scan: one thread per (batch, channel); 16-float state in VGPRs.
// Fuses +x*D and *silu(z); writes y in place over x_conv (same-thread RAW-safe).
__global__ __launch_bounds__(256)
void scan_kernel(float* __restrict__ xconv_y,
                 const float* __restrict__ xdbl,     // row stride DIN+2*DSTATE
                 const float* __restrict__ dlt,
                 const float* __restrict__ xz,       // z = cols [DIN, 2*DIN)
                 const float* __restrict__ A_log,
                 const float* __restrict__ Dvec) {
    const int c = blockIdx.x * 256 + threadIdx.x;
    const int b = blockIdx.y;

    float Acoef[DSTATE];
    #pragma unroll
    for (int s = 0; s < DSTATE; ++s) Acoef[s] = -__expf(A_log[c * DSTATE + s]);
    const float dval = Dvec[c];

    float st[DSTATE];
    #pragma unroll
    for (int s = 0; s < DSTATE; ++s) st[s] = 0.0f;

    for (int t = 0; t < SEQ; ++t) {
        const size_t row = (size_t)b * SEQ + t;
        const float dt = dlt[row * DIN + c];
        const float xv = xconv_y[row * DIN + c];
        const float zv = xz[row * (2 * DIN) + DIN + c];
        const float* bc = xdbl + row * (DIN + 2 * DSTATE) + DIN;  // B_t then C_t

        float Bt[DSTATE], Ct[DSTATE];
        #pragma unroll
        for (int q = 0; q < 4; ++q) {      // wave-uniform addresses -> scalarizable
            float4 bb = reinterpret_cast<const float4*>(bc)[q];
            float4 cc = reinterpret_cast<const float4*>(bc + DSTATE)[q];
            Bt[q*4+0] = bb.x; Bt[q*4+1] = bb.y; Bt[q*4+2] = bb.z; Bt[q*4+3] = bb.w;
            Ct[q*4+0] = cc.x; Ct[q*4+1] = cc.y; Ct[q*4+2] = cc.z; Ct[q*4+3] = cc.w;
        }

        float y = xv * dval;
        #pragma unroll
        for (int s = 0; s < DSTATE; ++s) {
            const float decay = __expf(dt * Acoef[s]);
            st[s] = fmaf(decay, st[s], Bt[s] * xv);
            y = fmaf(st[s], Ct[s], y);
        }
        y *= zv / (1.0f + __expf(-zv));    // * silu(z)
        xconv_y[row * DIN + c] = y;
    }
}

extern "C" void kernel_launch(void* const* d_in, const int* in_sizes, int n_in,
                              void* d_out, int out_size, void* d_ws, size_t ws_size,
                              hipStream_t stream) {
    (void)in_sizes; (void)n_in; (void)out_size; (void)ws_size;
    const float* x      = (const float*)d_in[0];
    const float* W_in   = (const float*)d_in[1];
    const float* w_conv = (const float*)d_in[2];
    const float* b_conv = (const float*)d_in[3];
    const float* W_x    = (const float*)d_in[4];
    const float* W_dt   = (const float*)d_in[5];
    const float* b_dt   = (const float*)d_in[6];
    const float* A_log  = (const float*)d_in[7];
    const float* Dv     = (const float*)d_in[8];
    const float* W_out  = (const float*)d_in[9];
    float* out = (float*)d_out;

    // Workspace layout (floats): xz | xconv(->y) | xdbl | delta  (~168 MB)
    float* xz    = (float*)d_ws;                          // 4096*4096
    float* xconv = xz    + (size_t)MROWS * (2 * DIN);     // 4096*2048
    float* xdbl  = xconv + (size_t)MROWS * DIN;           // 4096*2080
    float* dlt   = xdbl  + (size_t)MROWS * (DIN + 2 * DSTATE); // 4096*2048

    const dim3 blk(256);
    const int NDBL = DIN + 2 * DSTATE;                    // 2080

    // 1) xz = x @ W_in^T            (4096 x 4096 x 1024)
    gemm_bf16_wmma<0><<<dim3((2 * DIN + 255) / 256, MROWS / 128), blk, 0, stream>>>(
        x, DMODEL, W_in, DMODEL, xz, 2 * DIN, 2 * DIN, DMODEL, nullptr);

    // 2) x_conv = silu(causal_conv(x_inner) + b)
    conv_silu_kernel<<<dim3((MROWS * DIN) / 256), blk, 0, stream>>>(
        xz, w_conv, b_conv, xconv);

    // 3) x_dbl = x_conv @ W_x^T     (4096 x 2080 x 2048)
    gemm_bf16_wmma<0><<<dim3((NDBL + 255) / 256, MROWS / 128), blk, 0, stream>>>(
        xconv, DIN, W_x, DIN, xdbl, NDBL, NDBL, DIN, nullptr);

    // 4) delta = softplus(delta_in @ W_dt^T + b_dt)   (4096 x 2048 x 2048)
    gemm_bf16_wmma<1><<<dim3((DIN + 255) / 256, MROWS / 128), blk, 0, stream>>>(
        xdbl, NDBL, W_dt, DIN, dlt, DIN, DIN, DIN, b_dt);

    // 5) selective scan + x*D + silu(z) gating; y overwrites xconv
    scan_kernel<<<dim3(DIN / 256, BATCH), blk, 0, stream>>>(
        xconv, xdbl, dlt, xz, A_log, Dv);

    // 6) out = y @ W_out^T          (4096 x 1024 x 2048)
    gemm_bf16_wmma<0><<<dim3((DMODEL + 255) / 256, MROWS / 128), blk, 0, stream>>>(
        xconv, DIN, W_out, DIN, out, DMODEL, DMODEL, DIN, nullptr);
}
